// TransitionDown_32169305047425
// MI455X (gfx1250) — matrix operand order repaired and checked by reading, not compiled, not run on hardware
//
#include <hip/hip_runtime.h>
#include <hip/hip_bf16.h>
#include <float.h>

#define NPTS   32768
#define C_IN   64
#define C_OUT  128
#define KNN_K  16
#define MPTS   8192
#define FAN_IN 67
#define BN_EPS 1e-5f

typedef __attribute__((ext_vector_type(16))) __bf16 v16bf;
typedef __attribute__((ext_vector_type(8)))  float  v8f;

// ---------------------------------------------------------------------------
// 1) FPS: one persistent 1024-thread workgroup. min_d lives in LDS (128KB of
//    the 320KB WGP budget). Per step: strided update of 32 points/thread,
//    wave shfl_xor argmax (tie -> smaller index, matching jnp.argmax), then a
//    32-entry cross-wave reduce. Two barriers per step.
// ---------------------------------------------------------------------------
__global__ __launch_bounds__(1024)
void fps_kernel(const float* __restrict__ pos, float* __restrict__ subpos)
{
    extern __shared__ float md[];          // NPTS floats (dynamic LDS, 128KB)
    __shared__ float swv[32];
    __shared__ int   swi[32];
    __shared__ int   ssel;

    const int tid  = threadIdx.x;
    const int lane = tid & 31;
    const int wid  = tid >> 5;

    for (int p = tid; p < NPTS; p += 1024) md[p] = FLT_MAX;
    if (tid == 0) { subpos[0] = pos[0]; subpos[1] = pos[1]; subpos[2] = pos[2]; }
    __syncthreads();

    int sel = 0;   // selected[0] == 0
    for (int t = 1; t < MPTS; ++t) {
        const float sx = pos[sel*3+0], sy = pos[sel*3+1], sz = pos[sel*3+2];
        float bestv = -1.0f;
        int   besti = 0;
        #pragma unroll 4
        for (int j = 0; j < NPTS/1024; ++j) {
            const int p = tid + j*1024;
            const float dx = pos[p*3+0]-sx, dy = pos[p*3+1]-sy, dz = pos[p*3+2]-sz;
            float m = md[p];
            m = fminf(m, dx*dx + dy*dy + dz*dz);
            md[p] = m;
            if (m > bestv) { bestv = m; besti = p; }   // first occurrence kept
        }
        #pragma unroll
        for (int off = 16; off >= 1; off >>= 1) {
            const float ov = __shfl_xor(bestv, off, 32);
            const int   oi = __shfl_xor(besti, off, 32);
            if (ov > bestv || (ov == bestv && oi < besti)) { bestv = ov; besti = oi; }
        }
        if (lane == 0) { swv[wid] = bestv; swi[wid] = besti; }
        __syncthreads();
        if (wid == 0) {
            bestv = swv[lane]; besti = swi[lane];
            #pragma unroll
            for (int off = 16; off >= 1; off >>= 1) {
                const float ov = __shfl_xor(bestv, off, 32);
                const int   oi = __shfl_xor(besti, off, 32);
                if (ov > bestv || (ov == bestv && oi < besti)) { bestv = ov; besti = oi; }
            }
            if (lane == 0) ssel = besti;
        }
        __syncthreads();
        sel = ssel;
        if (tid < 3) subpos[t*3 + tid] = pos[sel*3 + tid];
    }
}

// ---------------------------------------------------------------------------
// 2) KNN: one query per thread, pos streamed through 24KB LDS tiles
//    (broadcast reads). Threshold-guarded top-16 in registers: ~K*ln(N/K)
//    heap updates expected per query, so the hot loop is 8 VALU ops.
// ---------------------------------------------------------------------------
#define KNN_TILE 2048
__global__ __launch_bounds__(256)
void knn_kernel(const float* __restrict__ pos, const float* __restrict__ subpos,
                int* __restrict__ nbr)
{
    __shared__ float tx[KNN_TILE], ty[KNN_TILE], tz[KNN_TILE];
    const int q = blockIdx.x * 256 + threadIdx.x;
    const float qx = subpos[q*3+0], qy = subpos[q*3+1], qz = subpos[q*3+2];

    float bd[KNN_K]; int bi[KNN_K];
    #pragma unroll
    for (int k = 0; k < KNN_K; ++k) { bd[k] = FLT_MAX; bi[k] = 0; }
    float worst = FLT_MAX; int wslot = 0;

    for (int base = 0; base < NPTS; base += KNN_TILE) {
        __syncthreads();
        for (int i = threadIdx.x; i < KNN_TILE; i += 256) {
            tx[i] = pos[(base+i)*3+0];
            ty[i] = pos[(base+i)*3+1];
            tz[i] = pos[(base+i)*3+2];
        }
        __syncthreads();
        for (int i = 0; i < KNN_TILE; ++i) {
            const float dx = tx[i]-qx, dy = ty[i]-qy, dz = tz[i]-qz;
            const float d = dx*dx + dy*dy + dz*dz;
            if (d < worst) {                 // rare path
                bd[wslot] = d; bi[wslot] = base + i;
                worst = bd[0]; wslot = 0;
                #pragma unroll
                for (int k = 1; k < KNN_K; ++k)
                    if (bd[k] >= worst) { worst = bd[k]; wslot = k; }
            }
        }
    }
    #pragma unroll
    for (int k = 0; k < KNN_K; ++k) nbr[q*KNN_K + k] = bi[k];   // order-invariant downstream
}

// ---------------------------------------------------------------------------
// 3) Pre-pack W[out=128, in=67] (row-major) into WMMA bf16 B-fragment layout,
//    K padded 67 -> 96 (3 k-steps of 32). B 32x16 layout: element j -> K=j
//    (j<8) or K=j+8 (j>=8) for lanes 0-15; lanes 16-31 = K+8;
//    lane&15 = output column. Also zero the BN accumulators each launch.
// ---------------------------------------------------------------------------
__global__ __launch_bounds__(256)
void prep_kernel(const float* __restrict__ W, __bf16* __restrict__ Bpack,
                 float* __restrict__ bn_sum, float* __restrict__ bn_sq)
{
    const int e = blockIdx.x * 256 + threadIdx.x;      // 3*8*32*16 = 12288 total
    if (e < 3*8*32*16) {
        const int j  = e & 15;
        const int ln = (e >> 4) & 31;
        const int n  = (e >> 9) & 7;
        const int s  = e >> 12;
        const int kk  = 32*s + j + ((j >= 8) ? 8 : 0) + ((ln >= 16) ? 8 : 0);
        const int col = 16*n + (ln & 15);
        const float v = (kk < FAN_IN) ? W[col*FAN_IN + kk] : 0.0f;
        Bpack[e] = (__bf16)v;
    }
    if (e < C_OUT) { bn_sum[e] = 0.0f; bn_sq[e] = 0.0f; }
}

// ---------------------------------------------------------------------------
// 4) Fused gather + GEMM + BN-stats + max-over-K. One wave per cluster:
//    the cluster's K=16 neighbor rows ARE the 16-row WMMA tile. A-fragment
//    gathered straight into the 16-bit A layout (lane<16: K={0..7,16..23},
//    lane>=16: +8). 8 N-tiles x 3 k-steps of v_wmma_f32_16x16x32_bf16.
//    Epilogue: BN needs only per-channel sums (float atomics, 128 slots) and
//    since rsqrt(var+eps)*gamma > 0 and ReLU is monotone, max-over-K commutes
//    with the affine+ReLU -> store the raw column max, never materialize h
//    (saves 2x64MB of traffic).
// ---------------------------------------------------------------------------
__global__ __launch_bounds__(256)
void gemm_kernel(const float* __restrict__ pos, const float* __restrict__ x,
                 const float* __restrict__ subpos, const int* __restrict__ nbr,
                 const v16bf* __restrict__ Bpack,
                 float* __restrict__ bn_sum, float* __restrict__ bn_sq,
                 float* __restrict__ hmax)
{
    const int lane = threadIdx.x & 31;
    const int m = (blockIdx.x * blockDim.x + threadIdx.x) >> 5;   // cluster id
    if (m >= MPTS) return;                                        // uniform per wave
    const int row = lane & 15;
    const int hi8 = (lane >> 4) << 3;                             // 0 or 8

    const int   r  = nbr[m*KNN_K + row];
    const float sx = subpos[m*3+0], sy = subpos[m*3+1], sz = subpos[m*3+2];

    v16bf a[3];
    #pragma unroll
    for (int s = 0; s < 3; ++s) {
        #pragma unroll
        for (int j = 0; j < 16; ++j) {
            const int F = 32*s + j + ((j >= 8) ? 8 : 0) + hi8;    // feature index
            float v = 0.0f;
            if (F < 3)           v = pos[r*3 + F] - ((F == 0) ? sx : (F == 1) ? sy : sz);
            else if (F < FAN_IN) v = x[r*C_IN + (F - 3)];
            a[s][j] = (__bf16)v;
        }
    }

    #pragma unroll
    for (int n = 0; n < 8; ++n) {
        v8f acc = {0.f, 0.f, 0.f, 0.f, 0.f, 0.f, 0.f, 0.f};
        #pragma unroll
        for (int s = 0; s < 3; ++s) {
            const v16bf b = Bpack[(s*8 + n)*32 + lane];
            acc = __builtin_amdgcn_wmma_f32_16x16x32_bf16(
                      false, a[s], false, b, (short)0, acc, false, false);
        }
        // lanes 0-15 hold rows 0-7 of channel c, lanes 16-31 rows 8-15
        float ssum = 0.f, ssq = 0.f, mx = -FLT_MAX;
        #pragma unroll
        for (int rr = 0; rr < 8; ++rr) {
            const float v = acc[rr];
            ssum += v; ssq += v*v; mx = fmaxf(mx, v);
        }
        ssum += __shfl_xor(ssum, 16, 32);
        ssq  += __shfl_xor(ssq,  16, 32);
        mx    = fmaxf(mx, __shfl_xor(mx, 16, 32));
        if (lane < 16) {
            const int c = 16*n + lane;
            atomicAdd(&bn_sum[c], ssum);
            atomicAdd(&bn_sq[c],  ssq);
            hmax[m*C_OUT + c] = mx;
        }
    }
}

// ---------------------------------------------------------------------------
// 5) Finalize: mean/var from the channel sums, affine + ReLU on the maxes.
// ---------------------------------------------------------------------------
__global__ __launch_bounds__(256)
void bn_out_kernel(const float* __restrict__ hmax, const float* __restrict__ bn_sum,
                   const float* __restrict__ bn_sq, const float* __restrict__ gamma,
                   const float* __restrict__ beta, float* __restrict__ xout)
{
    const int i = blockIdx.x * 256 + threadIdx.x;     // m*C_OUT + c
    const int c = i & (C_OUT - 1);
    const float inv_n = 1.0f / (float)(MPTS * KNN_K);
    const float mean  = bn_sum[c] * inv_n;
    const float var   = bn_sq[c] * inv_n - mean*mean;
    const float istd  = rsqrtf(var + BN_EPS);
    const float v = (hmax[i] - mean) * istd * gamma[c] + beta[c];
    xout[i] = fmaxf(v, 0.0f);
}

// ---------------------------------------------------------------------------
extern "C" void kernel_launch(void* const* d_in, const int* in_sizes, int n_in,
                              void* d_out, int out_size, void* d_ws, size_t ws_size,
                              hipStream_t stream)
{
    const float* pos   = (const float*)d_in[0];   // [N,3]
    const float* x     = (const float*)d_in[1];   // [N,64]
    const float* W     = (const float*)d_in[2];   // [128,67]
    const float* gamma = (const float*)d_in[3];   // [128]
    const float* beta  = (const float*)d_in[4];   // [128]

    float* out    = (float*)d_out;
    float* subpos = out;               // [M,3]   (tuple element 0)
    float* xout   = out + MPTS*3;      // [M,128] (tuple element 1)

    char* ws = (char*)d_ws;
    int*    nbr    = (int*)ws;                               // 512 KB
    __bf16* Bpack  = (__bf16*)(ws + (512<<10));              // 24 KB
    float*  bn_sum = (float*)(ws + (512<<10) + (32<<10));    // 512 B
    float*  bn_sq  = bn_sum + C_OUT;                         // 512 B
    float*  hmax   = (float*)(ws + (512<<10) + (36<<10));    // 4 MB

    prep_kernel  <<<48, 256, 0, stream>>>(W, Bpack, bn_sum, bn_sq);
    fps_kernel   <<<1, 1024, NPTS*sizeof(float), stream>>>(pos, subpos);
    knn_kernel   <<<MPTS/256, 256, 0, stream>>>(pos, subpos, nbr);
    gemm_kernel  <<<(MPTS*32)/256, 256, 0, stream>>>(pos, x, subpos, nbr,
                                                     (const v16bf*)Bpack,
                                                     bn_sum, bn_sq, hmax);
    bn_out_kernel<<<(MPTS*C_OUT)/256, 256, 0, stream>>>(hmax, bn_sum, bn_sq,
                                                        gamma, beta, xout);
}
